// GraphConv_74019466379559
// MI455X (gfx1250) — compile-verified, matrix operand order checked by reading, not run on hardware
//
#include <hip/hip_runtime.h>

#define NN 100000
#define EEDGE 3200000
#define EFD 8
#define FIN 8
#define FOUT 16
#define H1D 32
#define H2D 64
#define NEG_SLOPE 0.01f
#define BN_EPS 1e-5f

typedef __attribute__((ext_vector_type(16))) _Float16 v16h;
typedef __attribute__((ext_vector_type(8)))  _Float16 v8h;
typedef __attribute__((ext_vector_type(8)))  float    v8f;

// Fast sigmoid: v_exp_f32 + v_add_f32 + v_rcp_f32 (avoid the IEEE-division
// expansion: v_div_scale + 2x NR fma + fixup per element).
__device__ __forceinline__ float fast_sigmoid(float v) {
    float e = __expf(-v);                     // v_exp_f32 (log2e mul folded)
    return __builtin_amdgcn_rcpf(1.0f + e);   // v_rcp_f32
}

// Load a 16-half fragment in the CDNA5 16-bit WMMA operand layout.
// For operand row `row` with row stride `rowStride` halves:
//   lo = [kt*32 + 8*half + 0..7], hi = [kt*32 + 8*half + 16..23]
__device__ __forceinline__ v16h load_frag(const _Float16* base, int row,
                                          int rowStride, int kt, int half) {
    const _Float16* p = base + row * rowStride + kt * 32 + 8 * half;
    v8h lo = *(const v8h*)(p);
    v8h hi = *(const v8h*)(p + 16);
    v16h r;
#pragma unroll
    for (int i = 0; i < 8; ++i) { r[i] = lo[i]; r[8 + i] = hi[i]; }
    return r;
}

__device__ __forceinline__ v8f wmma_f16f32(v16h a, v16h b, v8f c) {
    return __builtin_amdgcn_wmma_f32_16x16x32_f16(
        /*neg_a=*/false, a, /*neg_b=*/false, b,
        /*c_mod=*/(short)0, c, /*reuse_a=*/false, /*reuse_b=*/false);
}

// ---------------------------------------------------------------------------
// Fused edge kernel: edge MLP (WMMA) -> per-edge bmm -> scatter-add atomics.
// 128 threads = 4 waves; each wave handles 4 tiles of 16 edges (256/block).
// ---------------------------------------------------------------------------
__global__ __launch_bounds__(128) void nnconv_edge_kernel(
    const float* __restrict__ x,
    const int*   __restrict__ ei,        // [2, E] int32 (src row, dst row)
    const float* __restrict__ ef,        // [E, 8]
    const float* __restrict__ W1, const float* __restrict__ b1,
    const float* __restrict__ W2, const float* __restrict__ b2,
    const float* __restrict__ W3, const float* __restrict__ b3,
    float* __restrict__ sums, float* __restrict__ cnt)
{
    // Weights transposed (N-major, K-contiguous) in f16 so B fragments are
    // two contiguous 16-byte LDS loads per lane.
    __shared__ __align__(16) _Float16 wT1[32 * 32];     // [n][k] K padded 8->32
    __shared__ __align__(16) _Float16 wT2[64 * 32];     // [n][k]
    __shared__ __align__(16) _Float16 wT3[128 * 64];    // [n][k]
    __shared__ float b1s[32];
    __shared__ float b2s[64];
    __shared__ float b3s[128];
    __shared__ __align__(16) _Float16 stA[4][16 * 32];  // per-wave L1 out
    __shared__ __align__(16) _Float16 stB[4][16 * 64];  // per-wave L2 out
    __shared__ __align__(16) _Float16 weS[4][16 * 128]; // per-wave We tile

    const int tid = threadIdx.x;

    for (int idx = tid; idx < 32 * 32; idx += 128) {
        int n = idx >> 5, k = idx & 31;
        wT1[idx] = (k < EFD) ? (_Float16)W1[k * H1D + n] : (_Float16)0.0f;
    }
    for (int idx = tid; idx < 64 * 32; idx += 128) {
        int n = idx >> 5, k = idx & 31;
        wT2[idx] = (_Float16)W2[k * H2D + n];
    }
    for (int idx = tid; idx < 128 * 64; idx += 128) {
        int n = idx >> 6, k = idx & 63;
        wT3[idx] = (_Float16)W3[k * (FIN * FOUT) + n];
    }
    if (tid < 32)  b1s[tid] = b1[tid];
    if (tid < 64)  b2s[tid] = b2[tid];
    if (tid < 128) b3s[tid] = b3[tid];
    __syncthreads();

    const int lane = tid & 31;
    const int wv   = tid >> 5;
    const int half = lane >> 4;   // 0: lanes 0-15, 1: lanes 16-31
    const int l15  = lane & 15;

    for (int t = 0; t < 4; ++t) {
        const int tile = blockIdx.x * 16 + wv * 4 + t;
        const int e0   = tile * 16;

        // ---- Layer 1 A fragment: edge features [16 x 8], K zero-padded ----
        v16h a1 = {};
        if (half == 0) {
            const int e = e0 + l15;
            if (e < EEDGE) {
                const float* p = ef + (long)e * EFD;
#pragma unroll
                for (int i = 0; i < EFD; ++i) a1[i] = (_Float16)p[i];
            }
        }

        // ---- Layer 1: [16x8] @ [8x32] + sigmoid -> stA ----
#pragma unroll
        for (int nt = 0; nt < 2; ++nt) {
            v16h b = load_frag(wT1, nt * 16 + l15, 32, 0, half);
            v8f c = {};
            c = wmma_f16f32(a1, b, c);
            const int   n  = nt * 16 + l15;
            const float bb = b1s[n];
#pragma unroll
            for (int g = 0; g < 8; ++g) {
                float v = fast_sigmoid(c[g] + bb);
                stA[wv][(g + 8 * half) * 32 + n] = (_Float16)v;
            }
        }
        asm volatile("" ::: "memory");

        // ---- Layer 2: [16x32] @ [32x64] + sigmoid -> stB ----
        v16h a2 = load_frag(stA[wv], l15, 32, 0, half);
#pragma unroll
        for (int nt = 0; nt < 4; ++nt) {
            v16h b = load_frag(wT2, nt * 16 + l15, 32, 0, half);
            v8f c = {};
            c = wmma_f16f32(a2, b, c);
            const int   n  = nt * 16 + l15;
            const float bb = b2s[n];
#pragma unroll
            for (int g = 0; g < 8; ++g) {
                float v = fast_sigmoid(c[g] + bb);
                stB[wv][(g + 8 * half) * 64 + n] = (_Float16)v;
            }
        }
        asm volatile("" ::: "memory");

        // ---- Layer 3: [16x64] @ [64x128] + bias -> weS (per-edge 8x16 We) ----
        v16h a3k0 = load_frag(stB[wv], l15, 64, 0, half);
        v16h a3k1 = load_frag(stB[wv], l15, 64, 1, half);
#pragma unroll
        for (int nt = 0; nt < 8; ++nt) {
            v8f c = {};
            c = wmma_f16f32(a3k0, load_frag(wT3, nt * 16 + l15, 64, 0, half), c);
            c = wmma_f16f32(a3k1, load_frag(wT3, nt * 16 + l15, 64, 1, half), c);
            const int   n  = nt * 16 + l15;
            const float bb = b3s[n];
#pragma unroll
            for (int g = 0; g < 8; ++g)
                weS[wv][(g + 8 * half) * 128 + n] = (_Float16)(c[g] + bb);
        }
        asm volatile("" ::: "memory");

        // ---- Per-edge bmm (x[src] 1x8 @ We 8x16) + scatter atomics ----
        // lane handles edge slot (lane&15), outputs [8*half, 8*half+8)
        const int e = e0 + l15;
        if (e < EEDGE) {
            const int src = ei[e];
            const int dst = ei[EEDGE + e];
            const float* xp = x + (long)src * FIN;
            const int ob = 8 * half;
            float acc[8];
#pragma unroll
            for (int oo = 0; oo < 8; ++oo) acc[oo] = 0.0f;
#pragma unroll
            for (int i = 0; i < FIN; ++i) {
                v8h w = *(const v8h*)&weS[wv][l15 * 128 + i * 16 + ob];
                const float xv = xp[i];
#pragma unroll
                for (int oo = 0; oo < 8; ++oo) acc[oo] += xv * (float)w[oo];
            }
#pragma unroll
            for (int oo = 0; oo < 8; ++oo)
                __hip_atomic_fetch_add(&sums[(long)dst * FOUT + ob + oo], acc[oo],
                                       __ATOMIC_RELAXED, __HIP_MEMORY_SCOPE_AGENT);
            if (half == 0)
                __hip_atomic_fetch_add(&cnt[dst], 1.0f,
                                       __ATOMIC_RELAXED, __HIP_MEMORY_SCOPE_AGENT);
        }
    }
}

// ---------------------------------------------------------------------------
__global__ void zero_kernel(float* __restrict__ p, long n) {
    long i = (long)blockIdx.x * blockDim.x + threadIdx.x;
    if (i < n) p[i] = 0.0f;
}

// scatter-mean finalize + x@root + bias + LeakyReLU; accumulate BN stats
__global__ __launch_bounds__(128) void node_pre_kernel(
    const float* __restrict__ x, const float* __restrict__ root,
    const float* __restrict__ bias, const float* __restrict__ sums,
    const float* __restrict__ cnt, float* __restrict__ pre,
    float* __restrict__ stats)
{
    __shared__ float ss[32];
    const int tid = threadIdx.x;
    if (tid < 32) ss[tid] = 0.0f;
    __syncthreads();

    const int n = blockIdx.x * 128 + tid;
    float ls[FOUT], lq[FOUT];
    if (n < NN) {
        const float inv = __builtin_amdgcn_rcpf(fmaxf(cnt[n], 1.0f));
        float xs[FIN];
#pragma unroll
        for (int i = 0; i < FIN; ++i) xs[i] = x[(long)n * FIN + i];
#pragma unroll
        for (int o = 0; o < FOUT; ++o) {
            float v = sums[(long)n * FOUT + o] * inv + bias[o];
#pragma unroll
            for (int i = 0; i < FIN; ++i) v += xs[i] * root[i * FOUT + o];
            v = (v >= 0.0f) ? v : NEG_SLOPE * v;   // LeakyReLU
            pre[(long)n * FOUT + o] = v;
            ls[o] = v;
            lq[o] = v * v;
        }
    } else {
#pragma unroll
        for (int o = 0; o < FOUT; ++o) { ls[o] = 0.0f; lq[o] = 0.0f; }
    }

#pragma unroll
    for (int o = 0; o < FOUT; ++o) {
        atomicAdd(&ss[o], ls[o]);          // ds_add_f32
        atomicAdd(&ss[16 + o], lq[o]);
    }
    __syncthreads();
    if (tid < 32)
        __hip_atomic_fetch_add(&stats[tid], ss[tid],
                               __ATOMIC_RELAXED, __HIP_MEMORY_SCOPE_AGENT);
}

// BatchNorm (training mode, biased variance) using accumulated stats
__global__ __launch_bounds__(128) void node_norm_kernel(
    const float* __restrict__ pre, const float* __restrict__ stats,
    const float* __restrict__ gamma, const float* __restrict__ beta,
    float* __restrict__ out)
{
    const int n = blockIdx.x * 128 + threadIdx.x;
    if (n >= NN) return;
#pragma unroll
    for (int o = 0; o < FOUT; ++o) {
        const float mean = stats[o] * (1.0f / (float)NN);
        const float var  = stats[16 + o] * (1.0f / (float)NN) - mean * mean;
        const float v = (pre[(long)n * FOUT + o] - mean) *
                        rsqrtf(var + BN_EPS) * gamma[o] + beta[o];
        out[(long)n * FOUT + o] = v;
    }
}

// ---------------------------------------------------------------------------
extern "C" void kernel_launch(void* const* d_in, const int* in_sizes, int n_in,
                              void* d_out, int out_size, void* d_ws, size_t ws_size,
                              hipStream_t stream) {
    (void)in_sizes; (void)n_in; (void)out_size; (void)ws_size;
    const float* x     = (const float*)d_in[0];
    const int*   ei    = (const int*)d_in[1];   // int64 in source downcasts to int32 under JAX defaults
    const float* ef    = (const float*)d_in[2];
    const float* W1    = (const float*)d_in[3];
    const float* b1    = (const float*)d_in[4];
    const float* W2    = (const float*)d_in[5];
    const float* b2    = (const float*)d_in[6];
    const float* W3    = (const float*)d_in[7];
    const float* b3    = (const float*)d_in[8];
    const float* root  = (const float*)d_in[9];
    const float* bias  = (const float*)d_in[10];
    const float* gamma = (const float*)d_in[11];
    const float* beta  = (const float*)d_in[12];
    float* out = (float*)d_out;

    // workspace layout (f32): sums[N*16] | cnt[N] | pre[N*16] | stats[32]
    float* sums  = (float*)d_ws;
    float* cnt   = sums + (long)NN * FOUT;
    float* pre   = cnt + NN;
    float* stats = pre + (long)NN * FOUT;

    const long zn = (long)NN * FOUT + NN;   // sums + cnt contiguous
    zero_kernel<<<(int)((zn + 255) / 256), 256, 0, stream>>>(sums, zn);
    zero_kernel<<<1, 32, 0, stream>>>(stats, 32);

    const int edgeBlocks = (EEDGE + 255) / 256;   // 256 edges per block, exact
    nnconv_edge_kernel<<<edgeBlocks, 128, 0, stream>>>(
        x, ei, ef, W1, b1, W2, b2, W3, b3, sums, cnt);

    node_pre_kernel<<<(NN + 127) / 128, 128, 0, stream>>>(
        x, root, bias, sums, cnt, pre, stats);
    node_norm_kernel<<<(NN + 127) / 128, 128, 0, stream>>>(
        pre, stats, gamma, beta, out);
}
